// MeshEdgeBlockConcat_79156247265436
// MI455X (gfx1250) — compile-verified, matrix operand order checked by reading, not run on hardware
//
#include <hip/hip_runtime.h>
#include <stdint.h>

// ---------------------------------------------------------------------------
// MeshGraphNet edge block on MI455X (gfx1250):
//   cat = [efeat | nfeat[src] | nfeat[dst]]  (E x 384)
//   h   = silu(cat @ w1 + b1)                (E x 128)
//   y   = LN(h @ w2 + b2) * gamma + beta + efeat
// bf16 WMMA (v_wmma_f32_16x16x32_bf16), f32 accumulate. Weights staged into
// LDS in per-lane fragment layout; layer1->layer2 transpose via LDS bf16.
// f32->bf16 packing via v_perm_b32 (1 VALU op per pair + rounding adds).
// ---------------------------------------------------------------------------

typedef __attribute__((ext_vector_type(16))) __bf16 v16bf;
typedef __attribute__((ext_vector_type(8)))  float  v8f;

#define DEDG   128
#define HDIM   128
#define WAVES  8
#define BLOCK  (WAVES * 32)
#define KT1    12          // 384 / 32 K-tiles for layer 1
#define KT2    4           // 128 / 32 K-tiles for layer 2
#define NT     8           // 128 / 16 N-tiles
#define HPAD   136         // h-scratch row stride in u16 (272B = 17*16B: keeps
                           // b128 alignment, spreads banks)

#define W1F_UINTS (KT1 * NT * 32 * 8)    // 24576 uints = 96 KB
#define W2F_UINTS (KT2 * NT * 32 * 8)    //  8192 uints = 32 KB
#define HSCR_U16  (WAVES * 16 * HPAD)    // 17408 u16  = 34 KB
#define SMEM_BYTES ((W1F_UINTS + W2F_UINTS) * 4 + HSCR_U16 * 2)

union BF16x16 {
    v16bf v;
    unsigned u[8];
    uint4 q[2];
};

// two f32 -> packed bf16 pair: one v_perm_b32 picking the high halves,
// after a +0x8000 round-to-nearest bias on each source.
__device__ __forceinline__ unsigned pack2bf(float a, float b) {
    unsigned ua = __float_as_uint(a) + 0x8000u;
    unsigned ub = __float_as_uint(b) + 0x8000u;
    // result bytes: [0]=ua.b2 [1]=ua.b3 [2]=ub.b2 [3]=ub.b3
    return __builtin_amdgcn_perm(ub, ua, 0x07060302u);
}

__device__ __forceinline__ unsigned short bf16bits(float a) {
    return (unsigned short)((__float_as_uint(a) + 0x8000u) >> 16);
}

__global__ __launch_bounds__(BLOCK) void
mesh_edge_block_wmma(const float* __restrict__ efeat,
                     const float* __restrict__ nfeat,
                     const int*   __restrict__ src,
                     const int*   __restrict__ dst,
                     const float* __restrict__ w1,
                     const float* __restrict__ b1,
                     const float* __restrict__ w2,
                     const float* __restrict__ b2,
                     const float* __restrict__ gamma,
                     const float* __restrict__ beta,
                     float* __restrict__ out,
                     int E, int nTiles) {
    extern __shared__ unsigned char smem_raw[];
    unsigned*       w1f  = (unsigned*)smem_raw;
    unsigned*       w2f  = w1f + W1F_UINTS;
    unsigned short* hscr = (unsigned short*)(w2f + W2F_UINTS);

    const int tid  = threadIdx.x;
    const int lane = tid & 31;
    const int wave = tid >> 5;
    const int row  = lane & 15;
    const int hi   = lane >> 4;

    // ---- Stage w1 into LDS as WMMA B fragments (bf16), once per block ----
    // B (32x16, K x N): lane<16 -> N=lane, K = kb..kb+7 (half0) / kb+16..kb+23
    //                   lane>=16 -> N=lane-16, K offset +8.
    for (int c = tid; c < KT1 * NT * 32 * 2; c += BLOCK) {
        int kt   = c >> 9;            // / (8*32*2)
        int rem  = c & 511;
        int nt   = rem >> 6;
        int ln   = (rem >> 1) & 31;
        int half = rem & 1;
        int n  = nt * 16 + (ln & 15);
        int kb = kt * 32 + ((ln >> 4) << 3) + half * 16;
        unsigned* dp = &w1f[(((kt << 3) + nt) * 32 + ln) * 8 + half * 4];
#pragma unroll
        for (int j = 0; j < 4; ++j) {
            float f0 = w1[(size_t)(kb + 2 * j)     * HDIM + n];
            float f1 = w1[(size_t)(kb + 2 * j + 1) * HDIM + n];
            dp[j] = pack2bf(f0, f1);
        }
    }
    // ---- Stage w2 ----
    for (int c = tid; c < KT2 * NT * 32 * 2; c += BLOCK) {
        int kt   = c >> 9;
        int rem  = c & 511;
        int nt   = rem >> 6;
        int ln   = (rem >> 1) & 31;
        int half = rem & 1;
        int n  = nt * 16 + (ln & 15);
        int kb = kt * 32 + ((ln >> 4) << 3) + half * 16;
        unsigned* dp = &w2f[(((kt << 3) + nt) * 32 + ln) * 8 + half * 4];
#pragma unroll
        for (int j = 0; j < 4; ++j) {
            float f0 = w2[(size_t)(kb + 2 * j)     * DEDG + n];
            float f1 = w2[(size_t)(kb + 2 * j + 1) * DEDG + n];
            dp[j] = pack2bf(f0, f1);
        }
    }
    __syncthreads();

    // ---- Per-lane constants (same for every tile) ----
    float b1v[NT], b2v[NT], gv[NT], bev[NT];
#pragma unroll
    for (int nt = 0; nt < NT; ++nt) {
        int N = nt * 16 + row;
        b1v[nt] = b1[N];
        b2v[nt] = b2[N];
        gv[nt]  = gamma[N];
        bev[nt] = beta[N];
    }

    const int tileStride = gridDim.x * WAVES;
    unsigned short* hb = hscr + wave * 16 * HPAD;

    // ---- Persistent loop: one 16-edge tile per wave per iteration ----
    for (int tile = blockIdx.x * WAVES + wave; tile < nTiles; tile += tileStride) {
        int e  = tile * 16 + row;
        int ec = e < E ? e : E - 1;          // clamp (loads stay uniform)
        int iS = src[ec];
        int iD = dst[ec];
        const float* rowE = efeat + (size_t)ec * DEDG;
        const float* rowS = nfeat + (size_t)iS * DEDG;
        const float* rowD = nfeat + (size_t)iD * DEDG;

        {   // prefetch next tile's efeat row (global_prefetch_b8)
            int ntile = tile + tileStride;
            if (ntile < nTiles) {
                int en = ntile * 16 + row;
                if (en < E) __builtin_prefetch(efeat + (size_t)en * DEDG, 0, 1);
            }
        }

        v8f acc[NT];
#pragma unroll
        for (int nt = 0; nt < NT; ++nt) acc[nt] = (v8f){0,0,0,0,0,0,0,0};

        // ======== Layer 1: cat[16x384] @ w1[384x128] ========
#pragma unroll
        for (int kt = 0; kt < KT1; ++kt) {
            const float* rp = (kt < 4) ? rowE : ((kt < 8) ? rowS : rowD);
            int cb = (kt & 3) * 32 + hi * 8;
            float4 f0 = *(const float4*)(rp + cb);
            float4 f1 = *(const float4*)(rp + cb + 4);
            float4 f2 = *(const float4*)(rp + cb + 16);
            float4 f3 = *(const float4*)(rp + cb + 20);
            BF16x16 a;
            a.u[0] = pack2bf(f0.x, f0.y); a.u[1] = pack2bf(f0.z, f0.w);
            a.u[2] = pack2bf(f1.x, f1.y); a.u[3] = pack2bf(f1.z, f1.w);
            a.u[4] = pack2bf(f2.x, f2.y); a.u[5] = pack2bf(f2.z, f2.w);
            a.u[6] = pack2bf(f3.x, f3.y); a.u[7] = pack2bf(f3.z, f3.w);
#pragma unroll
            for (int nt = 0; nt < NT; ++nt) {
                BF16x16 b;
                const uint4* bp = (const uint4*)&w1f[(((kt << 3) + nt) * 32 + lane) * 8];
                b.q[0] = bp[0]; b.q[1] = bp[1];
                acc[nt] = __builtin_amdgcn_wmma_f32_16x16x32_bf16(
                    false, a.v, false, b.v, (short)0, acc[nt], false, false);
            }
        }

        // ======== bias + SiLU, spill h to LDS (bf16, C-layout -> row-major) ====
#pragma unroll
        for (int nt = 0; nt < NT; ++nt) {
            float bias = b1v[nt];
            int N = nt * 16 + row;
#pragma unroll
            for (int r = 0; r < 8; ++r) {
                float x = acc[nt][r] + bias;
                float s = x / (1.0f + __expf(-x));     // silu
                hb[(r + hi * 8) * HPAD + N] = bf16bits(s);
            }
        }

        // ======== Layer 2: h[16x128] @ w2[128x128] ========
#pragma unroll
        for (int nt = 0; nt < NT; ++nt) acc[nt] = (v8f){0,0,0,0,0,0,0,0};
#pragma unroll
        for (int kt = 0; kt < KT2; ++kt) {
            int kb = kt * 32 + hi * 8;
            BF16x16 a;
            a.q[0] = *(const uint4*)(hb + row * HPAD + kb);
            a.q[1] = *(const uint4*)(hb + row * HPAD + kb + 16);
#pragma unroll
            for (int nt = 0; nt < NT; ++nt) {
                BF16x16 b;
                const uint4* bp = (const uint4*)&w2f[(((kt << 3) + nt) * 32 + lane) * 8];
                b.q[0] = bp[0]; b.q[1] = bp[1];
                acc[nt] = __builtin_amdgcn_wmma_f32_16x16x32_bf16(
                    false, a.v, false, b.v, (short)0, acc[nt], false, false);
            }
        }

        // ======== +b2, LayerNorm over N (row sums via half-wave shfl_xor) ====
#pragma unroll
        for (int nt = 0; nt < NT; ++nt) {
            float bias = b2v[nt];
#pragma unroll
            for (int r = 0; r < 8; ++r) acc[nt][r] += bias;
        }

        float mu[8], rs[8];
#pragma unroll
        for (int r = 0; r < 8; ++r) {
            float t1 = 0.f, t2 = 0.f;
#pragma unroll
            for (int nt = 0; nt < NT; ++nt) {
                float v = acc[nt][r];
                t1 += v;
                t2 += v * v;
            }
            // reduce across the 16-lane half (masks < 16 never cross halves)
            t1 += __shfl_xor(t1, 1);  t2 += __shfl_xor(t2, 1);
            t1 += __shfl_xor(t1, 2);  t2 += __shfl_xor(t2, 2);
            t1 += __shfl_xor(t1, 4);  t2 += __shfl_xor(t2, 4);
            t1 += __shfl_xor(t1, 8);  t2 += __shfl_xor(t2, 8);
            float m = t1 * (1.0f / 128.0f);
            float var = t2 * (1.0f / 128.0f) - m * m;
            mu[r] = m;
            rs[r] = rsqrtf(var + 1e-5f);
        }

        // ======== gamma/beta + residual + store ========
#pragma unroll
        for (int nt = 0; nt < NT; ++nt) {
            int N = nt * 16 + row;
#pragma unroll
            for (int r = 0; r < 8; ++r) {
                int M  = r + hi * 8;
                int eM = tile * 16 + M;
                if (eM < E) {
                    size_t addr = (size_t)eM * DEDG + N;
                    float yv = (acc[nt][r] - mu[r]) * rs[r] * gv[nt] + bev[nt]
                             + efeat[addr];
                    out[addr] = yv;
                }
            }
        }
    }
}

// Second output of the tuple: nfeat passthrough.
__global__ void copy_nodes_kernel(const float4* __restrict__ nf,
                                  float4* __restrict__ dstp, long long n4) {
    long long i = (long long)blockIdx.x * blockDim.x + threadIdx.x;
    if (i < n4) dstp[i] = nf[i];
}

extern "C" void kernel_launch(void* const* d_in, const int* in_sizes, int n_in,
                              void* d_out, int out_size, void* d_ws, size_t ws_size,
                              hipStream_t stream) {
    const float* efeat = (const float*)d_in[0];
    const float* nfeat = (const float*)d_in[1];
    const int*   src   = (const int*)d_in[2];
    const int*   dst   = (const int*)d_in[3];
    const float* w1    = (const float*)d_in[4];
    const float* b1    = (const float*)d_in[5];
    const float* w2    = (const float*)d_in[6];
    const float* b2    = (const float*)d_in[7];
    const float* gamma = (const float*)d_in[8];
    const float* beta  = (const float*)d_in[9];

    const int E = in_sizes[0] / DEDG;
    const int N = in_sizes[1] / DEDG;
    float* out = (float*)d_out;

    int nTiles = (E + 15) / 16;
    int blocks = 1024;
    if (blocks * WAVES > nTiles) blocks = (nTiles + WAVES - 1) / WAVES;
    if (blocks < 1) blocks = 1;

    mesh_edge_block_wmma<<<blocks, BLOCK, SMEM_BYTES, stream>>>(
        efeat, nfeat, src, dst, w1, b1, w2, b2, gamma, beta, out, E, nTiles);

    long long n4 = (long long)N * DEDG / 4;
    int cblocks = (int)((n4 + 255) / 256);
    copy_nodes_kernel<<<cblocks, 256, 0, stream>>>(
        (const float4*)nfeat, (float4*)(out + (size_t)E * DEDG), n4);
}
